// MultiheadAttention_32890859552884
// MI455X (gfx1250) — matrix operand order repaired and checked by reading, not compiled, not run on hardware
//
#include <hip/hip_runtime.h>
#include <hip/hip_fp16.h>

typedef __attribute__((ext_vector_type(16))) _Float16 v16h;
typedef __attribute__((ext_vector_type(8)))  float    v8f;
typedef __attribute__((ext_vector_type(4)))  _Float16 v4h;

constexpr int Bsz = 2, Sq = 2048, Dm = 1024, Hh = 16, HD = 64;

// compile-time unroll helper: gives a constexpr index so WMMA immediate
// modifier bits (reuse_a/reuse_b) can be frontend constants.
template <int I> struct IC { static constexpr int value = I; };
template <int N, int I = 0, typename F>
__device__ __forceinline__ void static_for(F&& f) {
  if constexpr (I < N) {
    f(IC<I>{});
    static_for<N, I + 1>(f);
  }
}

// ---------------------------------------------------------------------------
// f32 -> f16 conversion (4 elements / thread)
// ---------------------------------------------------------------------------
__global__ __launch_bounds__(256) void cvt_f32_f16(const float* __restrict__ in,
                                                   _Float16* __restrict__ out, int n) {
  int i = (blockIdx.x * 256 + threadIdx.x) * 4;
  if (i + 3 < n) {
    float4 f = *(const float4*)(in + i);
    v4h h = {(_Float16)f.x, (_Float16)f.y, (_Float16)f.z, (_Float16)f.w};
    *(v4h*)(out + i) = h;
  }
}

// ---------------------------------------------------------------------------
// C[M,N] = A[M,K] @ W[N,K]^T + bias[N]    (NT gemm, f16 in, f32 acc)
// 256 threads (8 waves), tile 128x128, K-step 64, double-buffered LDS,
// software-pipelined global loads, one barrier per K-step.
// ---------------------------------------------------------------------------
template <bool OUT_F16>
__global__ __launch_bounds__(256) void gemm_nt_bias(const _Float16* __restrict__ A,
                                                    const _Float16* __restrict__ W,
                                                    const float* __restrict__ bias,
                                                    void* __restrict__ Cv,
                                                    int M, int N, int K) {
  __shared__ _Float16 As[2][128][64];   // 32 KB
  __shared__ _Float16 Ws[2][128][64];   // 32 KB
  const int tid = threadIdx.x, wave = tid >> 5, lane = tid & 31;
  const int m0 = blockIdx.x * 128, n0 = blockIdx.y * 128;

  v8f acc[8];
  v8f zero = {};
#pragma unroll
  for (int j = 0; j < 8; ++j) acc[j] = zero;

  // prologue: tile 0 -> buffer 0 (128x64 halves per matrix, 4 x 16B / thread)
#pragma unroll
  for (int i = 0; i < 4; ++i) {
    int c = tid + i * 256;
    int r = c >> 3, off = (c & 7) * 8;
    *(uint4*)&As[0][r][off] = *(const uint4*)&A[(size_t)(m0 + r) * K + off];
    *(uint4*)&Ws[0][r][off] = *(const uint4*)&W[(size_t)(n0 + r) * K + off];
  }
  __syncthreads();

  const int nK = K / 64;
  for (int ks = 0; ks < nK; ++ks) {
    const int cur = ks & 1, nxt = cur ^ 1;

    // stage next tile into registers (overlaps with WMMA below)
    uint4 ra[4], rw[4];
    if (ks + 1 < nK) {
      const int k0 = (ks + 1) * 64;
#pragma unroll
      for (int i = 0; i < 4; ++i) {
        int c = tid + i * 256;
        int r = c >> 3, off = (c & 7) * 8;
        ra[i] = *(const uint4*)&A[(size_t)(m0 + r) * K + k0 + off];
        rw[i] = *(const uint4*)&W[(size_t)(n0 + r) * K + k0 + off];
      }
    }

    // compute: A fragment held constant across 8 WMMAs -> reuse_a hint
#pragma unroll
    for (int half = 0; half < 2; ++half) {
      v16h af;
      {
        unsigned int* ap = (unsigned int*)&af;
        const int am = wave * 16 + (lane & 15);
        const int akb = (lane >> 4) * 8 + half * 32;
#pragma unroll
        for (int v = 0; v < 8; ++v) {
          int k = ((v < 4) ? 2 * v : 16 + 2 * (v - 4)) + akb;
          ap[v] = *(const unsigned int*)&As[cur][am][k];
        }
      }
      static_for<8>([&](auto jc) {
        constexpr int j = decltype(jc)::value;
        v16h bf;
        unsigned int* bp = (unsigned int*)&bf;
        const int bn = j * 16 + (lane & 15);
        const int bkb = (lane >> 4) * 8 + half * 32;
#pragma unroll
        for (int v = 0; v < 8; ++v)
          bp[v] = *(const unsigned int*)&Ws[cur][bn][((v < 4) ? 2 * v : 16 + 2 * (v - 4)) + bkb];
        acc[j] = __builtin_amdgcn_wmma_f32_16x16x32_f16(false, af, false, bf,
                                                        (short)0, acc[j],
                                                        /*reuse_a=*/(j > 0), false);
      });
    }

    // drain staged registers into the other buffer
    if (ks + 1 < nK) {
#pragma unroll
      for (int i = 0; i < 4; ++i) {
        int c = tid + i * 256;
        int r = c >> 3, off = (c & 7) * 8;
        *(uint4*)&As[nxt][r][off] = ra[i];
        *(uint4*)&Ws[nxt][r][off] = rw[i];
      }
    }
    __syncthreads();
  }

#pragma unroll
  for (int j = 0; j < 8; ++j) {
    int col = n0 + j * 16 + (lane & 15);
    float bv = bias[col];
#pragma unroll
    for (int r = 0; r < 8; ++r) {
      int row = m0 + wave * 16 + (lane >> 4) * 8 + r;
      float val = acc[j][r] + bv;
      if (OUT_F16)
        ((_Float16*)Cv)[(size_t)row * N + col] = (_Float16)val;
      else
        ((float*)Cv)[(size_t)row * N + col] = val;
    }
  }
}

// ---------------------------------------------------------------------------
// Flash attention: grid (B*H, S/128), block 256 (8 waves, 16 q-rows / wave)
// Double-buffered K/V tiles, software-pipelined; online softmax in WMMA
// C-layout; P staged via per-wave LDS (same-wave DS ops are in-order, so no
// block barrier needed for the re-layout).
// ---------------------------------------------------------------------------
__global__ __launch_bounds__(256) void flash_attn(const _Float16* __restrict__ Qm,
                                                  const _Float16* __restrict__ Km,
                                                  const _Float16* __restrict__ Vm,
                                                  const float* __restrict__ mask,
                                                  _Float16* __restrict__ O) {
  __shared__ _Float16 Qs[128][64];        // 16 KB
  __shared__ _Float16 Ks[2][64][64];      // 16 KB  [key][hd]
  __shared__ _Float16 Vs[2][64][64];      // 16 KB  transposed: [hd][kv]
  __shared__ _Float16 Ps[8][16][64];      // 16 KB  per-wave P staging

  const int tid = threadIdx.x, wave = tid >> 5, lane = tid & 31;
  const int b = blockIdx.x >> 4, h = blockIdx.x & 15;
  const int q0 = blockIdx.y * 128;
  const size_t rowbase = (size_t)b * Sq;
  const int hc = h * HD;

  // ---- load Q tile (128x64) ----
#pragma unroll
  for (int i = 0; i < 4; ++i) {
    int c = tid + i * 256;
    int row = c >> 3, off = (c & 7) * 8;
    *(uint4*)&Qs[row][off] = *(const uint4*)&Qm[(rowbase + q0 + row) * Dm + hc + off];
  }
  // ---- load K/V tile 0 (V transposed into [hd][kv]) ----
#pragma unroll
  for (int i = 0; i < 2; ++i) {
    int c = tid + i * 256;
    int row = c >> 3, off = (c & 7) * 8;
    *(uint4*)&Ks[0][row][off] = *(const uint4*)&Km[(rowbase + row) * Dm + hc + off];
    uint4 vv = *(const uint4*)&Vm[(rowbase + row) * Dm + hc + off];
    alignas(16) _Float16 tmp[8];
    *(uint4*)tmp = vv;
#pragma unroll
    for (int e = 0; e < 8; ++e) Vs[0][off + e][row] = tmp[e];
  }
  __syncthreads();

  // Q A-fragments (held for the whole kv loop)
  v16h qa[2];
  {
    const int am = wave * 16 + (lane & 15), akb = (lane >> 4) * 8;
#pragma unroll
    for (int half = 0; half < 2; ++half) {
      unsigned int* qp = (unsigned int*)&qa[half];
#pragma unroll
      for (int v = 0; v < 8; ++v) {
        int k = ((v < 4) ? 2 * v : 16 + 2 * (v - 4)) + akb + half * 32;
        qp[v] = *(const unsigned int*)&Qs[am][k];
      }
    }
  }

  float mrow[8], lrow[8];
  v8f o[4];
  v8f zero = {};
#pragma unroll
  for (int r = 0; r < 8; ++r) { mrow[r] = -1e30f; lrow[r] = 0.f; }
#pragma unroll
  for (int t = 0; t < 4; ++t) o[t] = zero;

  for (int it = 0; it < Sq / 64; ++it) {
    const int cur = it & 1, nxt = cur ^ 1;
    const int kv0 = it * 64;

    // stage next K/V tile into registers (overlaps with compute)
    uint4 rk[2], rv[2];
    if (it + 1 < Sq / 64) {
#pragma unroll
      for (int i = 0; i < 2; ++i) {
        int c = tid + i * 256;
        int row = c >> 3, off = (c & 7) * 8;
        rk[i] = *(const uint4*)&Km[(rowbase + kv0 + 64 + row) * Dm + hc + off];
        rv[i] = *(const uint4*)&Vm[(rowbase + kv0 + 64 + row) * Dm + hc + off];
      }
    }

    // ---- scores: S = Q(16x64) @ K^T -> 4 tiles; A held across 4 WMMAs ----
    v8f sc[4];
#pragma unroll
    for (int t = 0; t < 4; ++t) sc[t] = zero;
#pragma unroll
    for (int half = 0; half < 2; ++half) {
      static_for<4>([&](auto tc) {
        constexpr int t = decltype(tc)::value;
        v16h kb;
        unsigned int* kp = (unsigned int*)&kb;
        const int key = t * 16 + (lane & 15);
        const int kbse = half * 32 + (lane >> 4) * 8;
#pragma unroll
        for (int v = 0; v < 8; ++v)
          kp[v] = *(const unsigned int*)&Ks[cur][key][kbse + ((v < 4) ? 2 * v : 16 + 2 * (v - 4))];
        sc[t] = __builtin_amdgcn_wmma_f32_16x16x32_f16(false, qa[half], false, kb,
                                                       (short)0, sc[t],
                                                       /*reuse_a=*/(t > 0), false);
      });
    }

    // ---- scale + additive mask ----
    const float scale = 0.125f;  // 1/sqrt(64)
#pragma unroll
    for (int t = 0; t < 4; ++t) {
#pragma unroll
      for (int r = 0; r < 8; ++r) {
        int qrow = q0 + wave * 16 + (lane >> 4) * 8 + r;
        int col = kv0 + t * 16 + (lane & 15);
        sc[t][r] = sc[t][r] * scale + mask[((size_t)b * Sq + qrow) * Sq + col];
      }
    }

    // ---- online softmax (row spans 16 lanes of one half-wave) ----
#pragma unroll
    for (int r = 0; r < 8; ++r) {
      float mx = fmaxf(fmaxf(sc[0][r], sc[1][r]), fmaxf(sc[2][r], sc[3][r]));
#pragma unroll
      for (int d = 1; d < 16; d <<= 1) mx = fmaxf(mx, __shfl_xor(mx, d, 32));
      float mnew = fmaxf(mrow[r], mx);
      float alpha = __expf(mrow[r] - mnew);
      float rs = 0.f;
#pragma unroll
      for (int t = 0; t < 4; ++t) {
        float p = __expf(sc[t][r] - mnew);
        sc[t][r] = p;
        rs += p;
      }
#pragma unroll
      for (int d = 1; d < 16; d <<= 1) rs += __shfl_xor(rs, d, 32);
      lrow[r] = lrow[r] * alpha + rs;
      mrow[r] = mnew;
#pragma unroll
      for (int t = 0; t < 4; ++t) o[t][r] *= alpha;
    }

    // ---- stage P (C-layout -> LDS), reload as A-fragments (wave-local) ----
#pragma unroll
    for (int t = 0; t < 4; ++t)
#pragma unroll
      for (int r = 0; r < 8; ++r)
        Ps[wave][(lane >> 4) * 8 + r][t * 16 + (lane & 15)] = (_Float16)sc[t][r];

    v16h pa[2];
    {
      const int pm = lane & 15, pkb = (lane >> 4) * 8;
#pragma unroll
      for (int half = 0; half < 2; ++half) {
        unsigned int* pp = (unsigned int*)&pa[half];
#pragma unroll
        for (int v = 0; v < 8; ++v) {
          int k = ((v < 4) ? 2 * v : 16 + 2 * (v - 4)) + pkb + half * 32;
          pp[v] = *(const unsigned int*)&Ps[wave][pm][k];
        }
      }
    }

    // ---- O += P(16x64) @ V(64kv x 64hd); A held across 4 WMMAs ----
#pragma unroll
    for (int half = 0; half < 2; ++half) {
      static_for<4>([&](auto tc) {
        constexpr int t = decltype(tc)::value;
        v16h vb;
        unsigned int* vp = (unsigned int*)&vb;
        const int hd = t * 16 + (lane & 15);
        const int kvb = half * 32 + (lane >> 4) * 8;
#pragma unroll
        for (int v = 0; v < 8; ++v)
          vp[v] = *(const unsigned int*)&Vs[cur][hd][kvb + ((v < 4) ? 2 * v : 16 + 2 * (v - 4))];
        o[t] = __builtin_amdgcn_wmma_f32_16x16x32_f16(false, pa[half], false, vb,
                                                      (short)0, o[t],
                                                      /*reuse_a=*/(t > 0), false);
      });
    }

    // ---- drain staged registers into the other K/V buffer ----
    if (it + 1 < Sq / 64) {
#pragma unroll
      for (int i = 0; i < 2; ++i) {
        int c = tid + i * 256;
        int row = c >> 3, off = (c & 7) * 8;
        *(uint4*)&Ks[nxt][row][off] = rk[i];
        alignas(16) _Float16 tmp[8];
        *(uint4*)tmp = rv[i];
#pragma unroll
        for (int e = 0; e < 8; ++e) Vs[nxt][off + e][row] = tmp[e];
      }
    }
    __syncthreads();
  }

  // ---- finalize: divide by l, store f16 [B*S, D] (heads re-interleaved) ----
#pragma unroll
  for (int t = 0; t < 4; ++t) {
#pragma unroll
    for (int r = 0; r < 8; ++r) {
      int qrow = q0 + wave * 16 + (lane >> 4) * 8 + r;
      float val = o[t][r] / lrow[r];
      O[(rowbase + qrow) * Dm + hc + t * 16 + (lane & 15)] = (_Float16)val;
    }
  }
}

// ---------------------------------------------------------------------------
extern "C" void kernel_launch(void* const* d_in, const int* in_sizes, int n_in,
                              void* d_out, int out_size, void* d_ws, size_t ws_size,
                              hipStream_t stream) {
  const float* q    = (const float*)d_in[0];
  const float* k    = (const float*)d_in[1];
  const float* v    = (const float*)d_in[2];
  const float* mask = (const float*)d_in[3];
  const float* WQw  = (const float*)d_in[4];
  const float* WQb  = (const float*)d_in[5];
  const float* WKw  = (const float*)d_in[6];
  const float* WKb  = (const float*)d_in[7];
  const float* WVw  = (const float*)d_in[8];
  const float* WVb  = (const float*)d_in[9];
  const float* WOw  = (const float*)d_in[10];
  const float* WOb  = (const float*)d_in[11];

  const size_t NA = (size_t)Bsz * Sq * Dm;  // 4 Mi activations
  const size_t NW = (size_t)Dm * Dm;        // 1 Mi weights
  _Float16* ws = (_Float16*)d_ws;           // 64 MB total f16 workspace
  _Float16 *Xq = ws,       *Xk = Xq + NA, *Xv = Xk + NA;
  _Float16 *Wq = Xv + NA,  *Wk = Wq + NW, *Wv = Wk + NW, *Wo = Wv + NW;
  _Float16 *Qa = Wo + NW,  *Ka = Qa + NA, *Va = Ka + NA, *Oa = Va + NA;

  // f32 -> f16 conversions
  cvt_f32_f16<<<NA / 1024, 256, 0, stream>>>(q, Xq, (int)NA);
  cvt_f32_f16<<<NA / 1024, 256, 0, stream>>>(k, Xk, (int)NA);
  cvt_f32_f16<<<NA / 1024, 256, 0, stream>>>(v, Xv, (int)NA);
  cvt_f32_f16<<<NW / 1024, 256, 0, stream>>>(WQw, Wq, (int)NW);
  cvt_f32_f16<<<NW / 1024, 256, 0, stream>>>(WKw, Wk, (int)NW);
  cvt_f32_f16<<<NW / 1024, 256, 0, stream>>>(WVw, Wv, (int)NW);
  cvt_f32_f16<<<NW / 1024, 256, 0, stream>>>(WOw, Wo, (int)NW);

  const int M = Bsz * Sq;  // 4096
  dim3 gproj(M / 128, Dm / 128);  // (32, 8)
  gemm_nt_bias<true><<<gproj, 256, 0, stream>>>(Xq, Wq, WQb, Qa, M, Dm, Dm);
  gemm_nt_bias<true><<<gproj, 256, 0, stream>>>(Xk, Wk, WKb, Ka, M, Dm, Dm);
  gemm_nt_bias<true><<<gproj, 256, 0, stream>>>(Xv, Wv, WVb, Va, M, Dm, Dm);

  flash_attn<<<dim3(Bsz * Hh, Sq / 128), 256, 0, stream>>>(Qa, Ka, Va, mask, Oa);

  gemm_nt_bias<false><<<gproj, 256, 0, stream>>>(Oa, Wo, WOb, d_out, M, Dm, Dm);
}
